// CRF_80453327389212
// MI455X (gfx1250) — matrix operand order repaired and checked by reading, not compiled, not run on hardware
//
#include <hip/hip_runtime.h>
#include <hip/hip_bf16.h>

typedef __attribute__((ext_vector_type(16))) _Float16 v16h;
typedef __attribute__((ext_vector_type(8)))  float    v8f;
typedef __attribute__((ext_vector_type(4)))  int      v4i_;

#define CRF_T 64
#define CRF_S 512
#define SA    68        // padded LDS row stride in floats (16B-aligned, conflict-free)
#define LOGB  5.0f      // >= log(sum_i exp(trans)): log(64)+0.1 = 4.26, + safety

// ---- async Global->LDS (ASYNCcnt) helpers ---------------------------------
__device__ __forceinline__ void async_copy_b128(const float* gsrc, float* ldst) {
#if __has_builtin(__builtin_amdgcn_global_load_async_to_lds_b128)
  float* gs = const_cast<float*>(gsrc);
  __builtin_amdgcn_global_load_async_to_lds_b128(
      (__attribute__((address_space(1))) v4i_*)gs,
      (__attribute__((address_space(3))) v4i_*)ldst, 0, 0);
#else
  const unsigned loff = (unsigned)(uintptr_t)ldst;   // low 32 bits == LDS offset
  asm volatile("global_load_async_to_lds_b128 %0, %1, off"
               :: "v"(loff), "v"(gsrc) : "memory");
#endif
}
template <int N>
__device__ __forceinline__ void wait_async() {
#if __has_builtin(__builtin_amdgcn_s_wait_asynccnt)
  __builtin_amdgcn_s_wait_asynccnt(N);
#else
  asm volatile("s_wait_asynccnt %0" :: "i"(N) : "memory");
#endif
}

// ---------------------------------------------------------------------------
// Forward (normalizer): block = 128 threads = 4 waves, 16 batches per block.
// Recurrent state: v[b][i] = exp(alpha - C) in LDS (f32) + offset C[b].
//   W (16b x 16j per wave) = V x exp(Trans)   [2x v_wmma_f32_16x16x32_f16]
//   m~ = max_j e_t[j] + LOGB ;  v' = exp(e_t - m~) * W ;  C += m~ (mask-gated)
// Emissions tiles stream in via triple-buffered GLOBAL_LOAD_ASYNC_TO_LDS,
// two steps ahead (each DMA overlaps ~2 iterations of compute).
// ---------------------------------------------------------------------------
__global__ __launch_bounds__(128) void crf_forward_kernel(
    const float* __restrict__ emissions,    // [B][S][T]
    const int*   __restrict__ mask,         // [B][S]
    const float* __restrict__ transitions,  // [T][T]
    float* __restrict__ denom)              // [B]
{
  __shared__ float vS[16 * SA];       // normalized probs, f32
  __shared__ float eS[3][16 * SA];    // triple-buffered emissions tiles
  __shared__ int   maskS[16 * CRF_S]; // whole mask slab for this block (32KB)
  __shared__ float scrE[128];
  __shared__ float me[16];
  __shared__ float Carr[16];
  __shared__ int   mrow[16];

  const int l2    = threadIdx.x;       // 0..127
  const int wv    = l2 >> 5;           // wave id = j-tile (0..3)
  const int lane  = l2 & 31;
  const int m     = lane & 15;
  const int hi    = lane >> 4;
  const int bbase = blockIdx.x * 16;

  // staging assignment: 8 lanes per batch row, 8 floats (32B) per lane
  const int bl = l2 >> 3;              // 0..15
  const int j0 = (l2 & 7) * 8;         // 0..56

  const float* erow = emissions + ((size_t)(bbase + bl) * CRF_S) * CRF_T + j0;

  // ---- pipeline prime: DMA tiles for t=1 and t=2
  async_copy_b128(erow + 1 * CRF_T,     &eS[1][bl * SA + j0]);
  async_copy_b128(erow + 1 * CRF_T + 4, &eS[1][bl * SA + j0 + 4]);
  async_copy_b128(erow + 2 * CRF_T,     &eS[2][bl * SA + j0]);
  async_copy_b128(erow + 2 * CRF_T + 4, &eS[2][bl * SA + j0 + 4]);

  // ---- resident B operand for this wave's j-tile: P = exp(transitions)
  // B (32x16 f16): element (k,n): lane = n + 16*(k/16), half-slot = k%16.
  v16h Bop0, Bop1;
  {
    const int j = wv * 16 + m;
#pragma unroll
    for (int s = 0; s < 16; ++s) {
      const int kl = hi * 16 + s;
      Bop0[s] = (_Float16)__expf(transitions[(kl)      * CRF_T + j]);
      Bop1[s] = (_Float16)__expf(transitions[(32 + kl) * CRF_T + j]);
    }
  }

  // ---- one-time mask preload (16 contiguous rows -> fully coalesced int4)
  {
    const int4* msrc = (const int4*)(mask + (size_t)bbase * CRF_S);
    int4* mdst = (int4*)maskS;
    for (int i = l2; i < 16 * CRF_S / 4; i += 128) mdst[i] = msrc[i];
  }

  // ---- init from e0 (synchronous): m0 = max e0, C = m0, v = exp(e0 - m0)
  float4 ea = *(const float4*)(erow);
  float4 eb = *(const float4*)(erow + 4);
  {
    float pm = fmaxf(fmaxf(fmaxf(ea.x, ea.y), fmaxf(ea.z, ea.w)),
                     fmaxf(fmaxf(eb.x, eb.y), fmaxf(eb.z, eb.w)));
    scrE[l2] = pm;
  }
  __syncthreads();
  if (l2 < 16) {
    float mm = scrE[l2 * 8];
#pragma unroll
    for (int q = 1; q < 8; ++q) mm = fmaxf(mm, scrE[l2 * 8 + q]);
    me[l2]   = mm;
    Carr[l2] = mm;
  }
  __syncthreads();
  {
    const float mb = me[bl];
    vS[bl * SA + j0 + 0] = __expf(ea.x - mb);
    vS[bl * SA + j0 + 1] = __expf(ea.y - mb);
    vS[bl * SA + j0 + 2] = __expf(ea.z - mb);
    vS[bl * SA + j0 + 3] = __expf(ea.w - mb);
    vS[bl * SA + j0 + 4] = __expf(eb.x - mb);
    vS[bl * SA + j0 + 5] = __expf(eb.y - mb);
    vS[bl * SA + j0 + 6] = __expf(eb.z - mb);
    vS[bl * SA + j0 + 7] = __expf(eb.w - mb);
  }
  wait_async<2>();     // tile t=1 landed (in-order); t=2 still in flight
  __syncthreads();

  // ---- main recursion
  for (int t = 1; t < CRF_S; ++t) {
    const int cur = t % 3;

    // issue DMA for t+2 (that buffer was last read in iteration t-1)
    if (t <= CRF_S - 3) {
      const int nb = (t + 2) % 3;
      async_copy_b128(erow + (size_t)(t + 2) * CRF_T,     &eS[nb][bl * SA + j0]);
      async_copy_b128(erow + (size_t)(t + 2) * CRF_T + 4, &eS[nb][bl * SA + j0 + 4]);
    }

    // A operand from vS (already exp'ed): 16-bit A 16x32 layout
    v16h A0, A1;
#pragma unroll
    for (int s = 0; s < 16; ++s) {
      const int kb = hi * 16 + s;
      A0[s] = (_Float16)vS[m * SA + kb];
      A1[s] = (_Float16)vS[m * SA + 32 + kb];
    }
    v8f acc = {};
    acc = __builtin_amdgcn_wmma_f32_16x16x32_f16(false, A0, false, Bop0,
                                                 (short)0, acc, false, false);
    acc = __builtin_amdgcn_wmma_f32_16x16x32_f16(false, A1, false, Bop1,
                                                 (short)0, acc, false, false);

    // per-lane partial max of current e tile (guaranteed resident)
    {
      float4 a = *(const float4*)&eS[cur][bl * SA + j0];
      float4 b = *(const float4*)&eS[cur][bl * SA + j0 + 4];
      scrE[l2] = fmaxf(fmaxf(fmaxf(a.x, a.y), fmaxf(a.z, a.w)),
                       fmaxf(fmaxf(b.x, b.y), fmaxf(b.z, b.w)));
    }
    __syncthreads();

    // m~ = max e + LOGB ; C += m~ (mask-gated); publish mask row
    if (l2 < 16) {
      float mm = scrE[l2 * 8];
#pragma unroll
      for (int q = 1; q < 8; ++q) mm = fmaxf(mm, scrE[l2 * 8 + q]);
      mm += LOGB;
      me[l2] = mm;
      const int mk = maskS[l2 * CRF_S + t];
      mrow[l2] = mk;
      if (mk) Carr[l2] += mm;
    }
    __syncthreads();

    // v' = exp(e - m~) * W (mask-selected). D layout: N=l&15, rows M=r+8*hi
    {
      const int j = wv * 16 + m;
#pragma unroll
      for (int r = 0; r < 8; ++r) {
        const int b    = r + hi * 8;
        const float vn = __expf(eS[cur][b * SA + j] - me[b]) * acc[r];
        const float vo = vS[b * SA + j];
        vS[b * SA + j] = mrow[b] ? vn : vo;
      }
    }
    // retire tile t+1 (needed next iteration); keep t+2 in flight
    if (t <= CRF_S - 3) wait_async<2>();
    else                wait_async<0>();
    __syncthreads();
  }

  // ---- final: denom[b] = C + log(sum_j v[b][j])
  {
    float ps = 0.0f;
#pragma unroll
    for (int q = 0; q < 8; ++q) ps += vS[bl * SA + j0 + q];
    scrE[l2] = ps;
    __syncthreads();
    if (l2 < 16) {
      float tot = scrE[l2 * 8];
#pragma unroll
      for (int q = 1; q < 8; ++q) tot += scrE[l2 * 8 + q];
      denom[bbase + l2] = Carr[l2] + __logf(tot);
    }
  }
}

// ---------------------------------------------------------------------------
// Numerator: per-batch gather/sum along the tagged path.
// ---------------------------------------------------------------------------
__global__ __launch_bounds__(256) void crf_numer_kernel(
    const float* __restrict__ emissions,
    const int*   __restrict__ tags,
    const int*   __restrict__ mask,
    const float* __restrict__ transitions,
    float* __restrict__ numer, int B)
{
  const int b = blockIdx.x * blockDim.x + threadIdx.x;
  if (b >= B) return;
  const float* eb = emissions + (size_t)b * CRF_S * CRF_T;
  const int*   tb = tags + (size_t)b * CRF_S;
  const int*   mb = mask + (size_t)b * CRF_S;
  int   prev = tb[0];
  float num  = eb[prev];
  for (int s = 1; s < CRF_S; ++s) {
    const int   tg = tb[s];
    const float mf = (float)mb[s];
    num += (eb[(size_t)s * CRF_T + tg] + transitions[tg * CRF_T + prev]) * mf;
    prev = tg;
  }
  numer[b] = num;
}

// ---------------------------------------------------------------------------
// Final mean over batch of (denom - numer).
// ---------------------------------------------------------------------------
__global__ __launch_bounds__(256) void crf_reduce_kernel(
    const float* __restrict__ denom, const float* __restrict__ numer,
    float* __restrict__ out, int B)
{
  __shared__ float sred[256];
  float s = 0.0f;
  for (int i = threadIdx.x; i < B; i += 256) s += denom[i] - numer[i];
  sred[threadIdx.x] = s;
  __syncthreads();
  for (int off = 128; off > 0; off >>= 1) {
    if (threadIdx.x < off) sred[threadIdx.x] += sred[threadIdx.x + off];
    __syncthreads();
  }
  if (threadIdx.x == 0) out[0] = sred[0] / (float)B;
}

extern "C" void kernel_launch(void* const* d_in, const int* in_sizes, int n_in,
                              void* d_out, int out_size, void* d_ws, size_t ws_size,
                              hipStream_t stream) {
  const float* emissions   = (const float*)d_in[0];
  const int*   tags        = (const int*)d_in[1];
  const int*   mask        = (const int*)d_in[2];
  const float* transitions = (const float*)d_in[3];

  const int B = in_sizes[1] / CRF_S;   // 1024

  float* denom = (float*)d_ws;
  float* numer = denom + B;

  crf_forward_kernel<<<B / 16, 128, 0, stream>>>(emissions, mask, transitions, denom);
  crf_numer_kernel<<<(B + 255) / 256, 256, 0, stream>>>(emissions, tags, mask,
                                                        transitions, numer, B);
  crf_reduce_kernel<<<1, 256, 0, stream>>>(denom, numer, (float*)d_out, B);
}
